// MyGPT2Attention_26302379721034
// MI455X (gfx1250) — matrix operand order repaired and checked by reading, not compiled, not run on hardware
//
#include <hip/hip_runtime.h>
#include <hip/hip_bf16.h>

// ---------------------------------------------------------------------------
// GPT-2 attention block on gfx1250 (CDNA5), all matmuls via v_wmma_f32_16x16x32_f16
//   B=4, S=2048, D=1024, H=16, HD=64
//   kernel 1: qkv[8192,3072] = X[8192,1024] @ W_attn + b_attn   (f16 out)
//   kernel 2: flash causal attention -> Obuf[8192,1024] f16
//   kernel 3: out[8192,1024] = Obuf @ W_proj + b_proj           (f32 out)
// CDNA5 data-movement paths used:
//   - global_load_async_to_lds_b128 (ASYNCcnt) for proj-GEMM A staging
//   - TENSOR_LOAD_TO_LDS (TDM, TENSORcnt) for the attention V tile, with
//     descriptor padding reproducing the Vs[64][72] LDS layout
//   - global_prefetch for the next K/V block
// ---------------------------------------------------------------------------

typedef __attribute__((ext_vector_type(16))) _Float16 v16h;
typedef __attribute__((ext_vector_type(8)))  float    v8f;
typedef __attribute__((ext_vector_type(4)))  unsigned int v4u;
typedef __attribute__((ext_vector_type(8)))  int      v8i_;
typedef __attribute__((ext_vector_type(4)))  int      v4i_;

#define NB   4
#define SS   2048
#define DD   1024
#define NH   16
#define HD   64
#define MM   (NB * SS)        // 8192 rows

#if defined(__AMDGCN__) && __has_builtin(__builtin_amdgcn_tensor_load_to_lds)
#define HAVE_TDM 1
#else
#define HAVE_TDM 0
#endif

// WMMA A-fragment K index for element i, half-wave khi (16-bit A 16x32 layout)
__device__ __forceinline__ int a_k(int i, int khi) {
    return i + (khi << 3) + ((i >= 8) ? 8 : 0);
}

__device__ __forceinline__ v8f wmma_f16(v16h a, v16h b, v8f c) {
    return __builtin_amdgcn_wmma_f32_16x16x32_f16(false, a, false, b,
                                                  (short)0, c, false, false);
}

// Generic pointers into LDS carry the wave-relative LDS byte offset in the
// low 32 bits (flat-LDS aperture addressing), so this yields the raw dsaddr.
__device__ __forceinline__ unsigned int lds_off(const void* p) {
    return (unsigned int)(unsigned long long)p;
}

// CDNA5 async global->LDS copy (per-lane 16B), tracked with ASYNCcnt.
__device__ __forceinline__ void async_ld_b128(unsigned int lds, const void* g) {
    asm volatile("global_load_async_to_lds_b128 %0, %1, off"
                 :: "v"(lds), "v"((unsigned long long)g) : "memory");
}
__device__ __forceinline__ void wait_async0() {
    asm volatile("s_wait_asynccnt 0x0" ::: "memory");
}

#if HAVE_TDM
// TDM: DMA a 2D f16 tile [64 rows x 64 cols], row stride 3072 elements, into
// LDS with +8-half row padding (pad 4 DWORDs after every 32 DWORDs), i.e. the
// exact Vs[64][72] layout. Issued once per wave (EXEC ignored by TDM).
__device__ __forceinline__ void tdm_load_64x64_f16(const void* gsrc,
                                                   unsigned int lds_dst) {
    const unsigned long long ga = (unsigned long long)gsrc;
    v4u g0;
    g0[0] = 1u;                                   // count=1 valid descriptor
    g0[1] = lds_dst;                              // lds_addr (bytes)
    g0[2] = (unsigned int)ga;                     // global_addr[31:0]
    g0[3] = (unsigned int)((ga >> 32) & 0x01FFFFFFull) | (2u << 30); // addr[56:32], type=2
    v8i_ g1;
    g1[0] = (1 << 16)      // data_size = 2 bytes
          | (1 << 20)      // pad_enable
          | (4 << 22)      // pad_interval: 32 DWORDs
          | (3 << 25);     // pad_amount:   4 DWORDs
    g1[1] = (64 << 16);    // tensor_dim0 = 64 (lo 16 bits in [31:16])
    g1[2] = (64 << 16);    // tensor_dim1 = 64
    g1[3] = (64 << 16);    // tile_dim0   = 64
    g1[4] = 64;            // tile_dim1   = 64, tile_dim2 = 0
    g1[5] = 3072;          // tensor_dim0_stride = 3072 elements
    g1[6] = 0;
    g1[7] = 0;
    v4i_ z4 = {};
#if __clang_major__ >= 23
    v8i_ z8 = {};
    __builtin_amdgcn_tensor_load_to_lds(g0, g1, z4, z4, z8, 0);
#else
    __builtin_amdgcn_tensor_load_to_lds(g0, g1, z4, z4, 0);
#endif
}
#endif

// ---------------------------------------------------------------------------
// Generic f16-WMMA GEMM:  C[M,N] = A[M,K] * B[K,N] + bias[N]
//   A: f32 (A_IS_HALF=false) or f16 (true), row-major, ld=K
//   B: f32 row-major, ld=N   (converted to f16 into LDS)
//   C: f16 (OUT_HALF=true) or f32, ld=N
// Workgroup: 256 threads (8 waves), tile 128(M) x 64(N), K-step 32.
// ---------------------------------------------------------------------------
template <bool A_IS_HALF, bool OUT_HALF>
__global__ __launch_bounds__(256)
void gemm_wmma_kernel(const void* __restrict__ Aptr,
                      const float* __restrict__ Bptr,
                      const float* __restrict__ bias,
                      void* __restrict__ Cptr,
                      int M, int N, int K) {
    __shared__ _Float16 Ash[128][32 + 8];
    __shared__ _Float16 Bsh[32][64 + 8];

    const int tid   = threadIdx.x;
    const int wave  = tid >> 5;
    const int lane  = tid & 31;
    const int mlo   = lane & 15;
    const int khi   = lane >> 4;
    const int waveM = wave & 3;      // 0..3
    const int waveN = wave >> 2;     // 0..1

    const int m_block = blockIdx.x * 128;
    const int n_block = blockIdx.y * 64;

    v8f c[2][2];
    v8f zero = {};
    c[0][0] = zero; c[0][1] = zero; c[1][0] = zero; c[1][1] = zero;

    for (int k0 = 0; k0 < K; k0 += 32) {
        __syncthreads();
        // ---- stage A tile: 128 rows x 32 k  (2 threads per row) ----
        {
            const int row = tid >> 1;
            const int kp  = (tid & 1) * 16;
            if constexpr (A_IS_HALF) {
                const _Float16* src = (const _Float16*)Aptr +
                                      (size_t)(m_block + row) * K + k0 + kp;
                const unsigned int dst = lds_off(&Ash[row][kp]);
                async_ld_b128(dst,      src);        // 8 halves
                async_ld_b128(dst + 16, src + 8);    // 8 halves
            } else {
                const float* src = (const float*)Aptr +
                                   (size_t)(m_block + row) * K + k0 + kp;
#pragma unroll
                for (int i = 0; i < 16; ++i) Ash[row][kp + i] = (_Float16)src[i];
            }
        }
        // ---- stage B tile: 32 rows x 64 n  (8 threads per row) ----
        {
            const int row = tid >> 3;
            const int np  = (tid & 7) * 8;
            const float* src = Bptr + (size_t)(k0 + row) * N + n_block + np;
#pragma unroll
            for (int i = 0; i < 8; ++i) Bsh[row][np + i] = (_Float16)src[i];
        }
        if constexpr (A_IS_HALF) wait_async0();
        __syncthreads();

        // ---- fragments + wmma ----
        v16h a[2], b[2];
#pragma unroll
        for (int mt = 0; mt < 2; ++mt) {
            const int mr = waveM * 32 + mt * 16 + mlo;
#pragma unroll
            for (int i = 0; i < 16; i += 2) {
                const int k = a_k(i, khi);
                a[mt][i]     = Ash[mr][k];
                a[mt][i + 1] = Ash[mr][k + 1];
            }
        }
#pragma unroll
        for (int nt = 0; nt < 2; ++nt) {
            const int nc = waveN * 32 + nt * 16;
#pragma unroll
            for (int i = 0; i < 16; ++i) b[nt][i] = Bsh[a_k(i, khi)][nc + mlo];
        }
#pragma unroll
        for (int mt = 0; mt < 2; ++mt)
#pragma unroll
            for (int nt = 0; nt < 2; ++nt)
                c[mt][nt] = wmma_f16(a[mt], b[nt], c[mt][nt]);
    }

    // ---- epilogue: bias + store ----
#pragma unroll
    for (int mt = 0; mt < 2; ++mt) {
#pragma unroll
        for (int nt = 0; nt < 2; ++nt) {
            const int ncol = n_block + waveN * 32 + nt * 16 + mlo;
            const float bv = bias[ncol];
#pragma unroll
            for (int j = 0; j < 8; ++j) {
                const int mrow = m_block + waveM * 32 + mt * 16 + j + (khi << 3);
                const float val = c[mt][nt][j] + bv;
                if constexpr (OUT_HALF)
                    ((_Float16*)Cptr)[(size_t)mrow * N + ncol] = (_Float16)val;
                else
                    ((float*)Cptr)[(size_t)mrow * N + ncol] = val;
            }
        }
    }
}

// ---------------------------------------------------------------------------
// Flash causal attention, one WG per (b*H+h, q-block of 64 queries).
// 128 threads = 4 waves; wave w owns queries [q0+16w, q0+16w+16).
// qkv layout: [B*S, 3072] f16; q at col h*64, k at 1024+h*64, v at 2048+h*64.
// ---------------------------------------------------------------------------
__global__ __launch_bounds__(128)
void attn_flash_kernel(const _Float16* __restrict__ qkv,
                       _Float16* __restrict__ Obuf) {
    __shared__ _Float16 KT[64][72];        // [hd][key]   (K transposed)
    __shared__ _Float16 Vs[64][72];        // [key][hd]
    __shared__ _Float16 Ps[4][16][72];     // per-wave P staging [qrow][key]

    const int bh  = blockIdx.x;            // b*H + h
    const int b   = bh >> 4;
    const int h   = bh & 15;
    const int qb  = blockIdx.y;
    const int q0  = qb * 64;

    const int tid  = threadIdx.x;
    const int wave = tid >> 5;
    const int lane = tid & 31;
    const int mlo  = lane & 15;
    const int khi  = lane >> 4;

    const size_t rs   = 3 * (size_t)DD;            // 3072 row stride
    const size_t base = (size_t)b * SS * rs;
    const int hoff    = h * HD;

    // ---- load Q fragments once (hd split into 2 chunks of 32) ----
    v16h qa[2];
    {
        const int mrow = q0 + wave * 16 + mlo;
        const _Float16* qrow = qkv + base + (size_t)mrow * rs + hoff;
#pragma unroll
        for (int cch = 0; cch < 2; ++cch)
#pragma unroll
            for (int i = 0; i < 16; i += 2) {
                const int k = a_k(i, khi);
                qa[cch][i]     = qrow[cch * 32 + k];
                qa[cch][i + 1] = qrow[cch * 32 + k + 1];
            }
    }

    v8f O[4];
    v8f zero = {};
#pragma unroll
    for (int t = 0; t < 4; ++t) O[t] = zero;
    float mr[8], lr[8];
#pragma unroll
    for (int j = 0; j < 8; ++j) { mr[j] = -1e30f; lr[j] = 0.0f; }

    for (int kb = 0; kb <= qb; ++kb) {
        const int k0 = kb * 64;
        __syncthreads();
#if HAVE_TDM
        // ---- V tile via Tensor Data Mover: one DMA issued by wave 0 ----
        if (wave == 0)
            tdm_load_64x64_f16(qkv + base + (size_t)k0 * rs + 2 * DD + hoff,
                               lds_off(&Vs[0][0]));
#endif
        // ---- stage K^T (VALU transpose); V fallback via async copies ----
        {
            const int key = tid >> 1;
            const int dp  = (tid & 1) * 32;
            const _Float16* krow = qkv + base + (size_t)(k0 + key) * rs + DD + hoff + dp;
            const _Float16* vrow = qkv + base + (size_t)(k0 + key) * rs + 2 * DD + hoff + dp;
#if !HAVE_TDM
            const unsigned int vdst = lds_off(&Vs[key][dp]);
#pragma unroll
            for (int i = 0; i < 4; ++i)
                async_ld_b128(vdst + i * 16, vrow + i * 8);
#endif
#pragma unroll
            for (int i = 0; i < 32; ++i) KT[dp + i][key] = krow[i];
            // prefetch next key block while we compute on this one
            if (kb < qb) {
                __builtin_prefetch(krow + 64 * rs, 0, 0);
                __builtin_prefetch(vrow + 64 * rs, 0, 0);
            }
        }
#if HAVE_TDM
        __builtin_amdgcn_s_wait_tensorcnt(0);
#else
        wait_async0();
#endif
        __syncthreads();

        // ---- scores: 4 key tiles of 16, each = 2 wmma over hd ----
        v8f st[4];
#pragma unroll
        for (int kc = 0; kc < 4; ++kc) {
            v8f acc = zero;
#pragma unroll
            for (int cch = 0; cch < 2; ++cch) {
                v16h kf;
#pragma unroll
                for (int i = 0; i < 16; ++i) kf[i] = KT[cch * 32 + a_k(i, khi)][kc * 16 + mlo];
                acc = wmma_f16(qa[cch], kf, acc);
            }
            st[kc] = acc;
        }

        // ---- scale + causal mask (diagonal block only) ----
        const bool diag = (kb == qb);
#pragma unroll
        for (int kc = 0; kc < 4; ++kc)
#pragma unroll
            for (int j = 0; j < 8; ++j) {
                float s = st[kc][j] * 0.125f;   // 1/sqrt(64)
                if (diag) {
                    const int ng = k0 + kc * 16 + mlo;
                    const int mg = q0 + wave * 16 + j + (khi << 3);
                    if (ng > mg) s = -1e30f;
                }
                st[kc][j] = s;
            }

        // ---- online softmax (rows live across 16-lane half-waves) ----
        float pexp[4][8];
#pragma unroll
        for (int j = 0; j < 8; ++j) {
            float rv = st[0][j];
#pragma unroll
            for (int kc = 1; kc < 4; ++kc) rv = fmaxf(rv, st[kc][j]);
#pragma unroll
            for (int off = 8; off >= 1; off >>= 1)
                rv = fmaxf(rv, __shfl_xor(rv, off, 32));
            const float mnew = fmaxf(mr[j], rv);
            const float corr = __expf(mr[j] - mnew);
            mr[j] = mnew;
            float rsum = 0.0f;
#pragma unroll
            for (int kc = 0; kc < 4; ++kc) {
                const float p = __expf(st[kc][j] - mnew);
                pexp[kc][j] = p;
                rsum += p;
            }
#pragma unroll
            for (int off = 8; off >= 1; off >>= 1)
                rsum += __shfl_xor(rsum, off, 32);
            lr[j] = lr[j] * corr + rsum;
#pragma unroll
            for (int t = 0; t < 4; ++t) O[t][j] *= corr;
        }

        // ---- transpose P through per-wave LDS into A-fragment layout ----
#pragma unroll
        for (int kc = 0; kc < 4; ++kc)
#pragma unroll
            for (int j = 0; j < 8; ++j)
                Ps[wave][j + (khi << 3)][kc * 16 + mlo] = (_Float16)pexp[kc][j];
        __syncthreads();

        // ---- O += P(16x64) @ V(64x64): 2 key chunks x 4 hd tiles ----
#pragma unroll
        for (int cch = 0; cch < 2; ++cch) {
            v16h pf;
#pragma unroll
            for (int i = 0; i < 16; i += 2) {
                const int k = a_k(i, khi);
                pf[i]     = Ps[wave][mlo][cch * 32 + k];
                pf[i + 1] = Ps[wave][mlo][cch * 32 + k + 1];
            }
#pragma unroll
            for (int t = 0; t < 4; ++t) {
                v16h vf;
#pragma unroll
                for (int i = 0; i < 16; ++i) vf[i] = Vs[cch * 32 + a_k(i, khi)][t * 16 + mlo];
                O[t] = wmma_f16(pf, vf, O[t]);
            }
        }
    }

    // ---- normalize + store f16 [B*S, D] ----
#pragma unroll
    for (int t = 0; t < 4; ++t)
#pragma unroll
        for (int j = 0; j < 8; ++j) {
            const int mrow = q0 + wave * 16 + j + (khi << 3);
            const float val = O[t][j] / lr[j];
            Obuf[((size_t)b * SS + mrow) * DD + hoff + t * 16 + mlo] = (_Float16)val;
        }
}

// ---------------------------------------------------------------------------
extern "C" void kernel_launch(void* const* d_in, const int* in_sizes, int n_in,
                              void* d_out, int out_size, void* d_ws, size_t ws_size,
                              hipStream_t stream) {
    const float* hidden = (const float*)d_in[0];  // [B,S,D]
    const float* W_attn = (const float*)d_in[1];  // [D,3D]
    const float* b_attn = (const float*)d_in[2];  // [3D]
    const float* W_proj = (const float*)d_in[3];  // [D,D]
    const float* b_proj = (const float*)d_in[4];  // [D]
    float* out = (float*)d_out;

    _Float16* qkv  = (_Float16*)d_ws;                                    // [8192,3072] f16 (48MB)
    _Float16* Obuf = (_Float16*)((char*)d_ws + (size_t)MM * 3 * DD * 2); // [8192,1024] f16 (16MB)

    // 1) qkv = X @ W_attn + b_attn
    {
        dim3 grid(MM / 128, (3 * DD) / 64);
        gemm_wmma_kernel<false, true><<<grid, 256, 0, stream>>>(
            hidden, W_attn, b_attn, (void*)qkv, MM, 3 * DD, DD);
    }
    // 2) flash causal attention
    {
        dim3 grid(NB * NH, SS / 64);
        attn_flash_kernel<<<grid, 128, 0, stream>>>(qkv, Obuf);
    }
    // 3) out = Obuf @ W_proj + b_proj
    {
        dim3 grid(MM / 128, DD / 64);
        gemm_wmma_kernel<true, false><<<grid, 256, 0, stream>>>(
            (void*)Obuf, W_proj, b_proj, (void*)out, MM, DD, DD);
    }
}
